// RATransformerEncoderLayer_70884140253548
// MI455X (gfx1250) — matrix operand order repaired and checked by reading, not compiled, not run on hardware
//
#include <hip/hip_runtime.h>
#include <hip/hip_bf16.h>

// ---------------------------------------------------------------------------
// Types for CDNA5 WMMA (wave32)
// ---------------------------------------------------------------------------
typedef __bf16 bf16_t;
typedef __attribute__((ext_vector_type(16))) __bf16 bf16x16;
typedef __attribute__((ext_vector_type(8)))  __bf16 bf16x8;
typedef __attribute__((ext_vector_type(8)))  float  f32x8;

static __device__ __forceinline__ bf16_t f2bf(float f) {
  unsigned u = __builtin_bit_cast(unsigned, f);
  unsigned r = u + 0x7FFFu + ((u >> 16) & 1u);   // round-to-nearest-even
  unsigned short h = (unsigned short)(r >> 16);
  return __builtin_bit_cast(bf16_t, h);
}

static __device__ __forceinline__ bf16x16 ld_frag(const bf16_t* p0, const bf16_t* p1) {
  bf16x8 a = *(const bf16x8*)p0;
  bf16x8 b = *(const bf16x8*)p1;
  bf16x16 r;
#pragma unroll
  for (int i = 0; i < 8; ++i) { r[i] = a[i]; r[i + 8] = b[i]; }
  return r;
}

// ---------------------------------------------------------------------------
// fp32 -> bf16 conversion (elementwise)
// ---------------------------------------------------------------------------
__global__ void cvt_f32_bf16(const float* __restrict__ in, bf16_t* __restrict__ out, int n) {
  int i = blockIdx.x * blockDim.x + threadIdx.x;
  if (i < n) out[i] = f2bf(in[i]);
}

// rel_v_emb [64,64] -> transposed bf16 [d][r]
__global__ void cvt_transpose64(const float* __restrict__ in, bf16_t* __restrict__ out) {
  int i = blockIdx.x * blockDim.x + threadIdx.x;   // 4096
  if (i < 4096) {
    int r = i >> 6, d = i & 63;
    out[d * 64 + r] = f2bf(in[r * 64 + d]);
  }
}

// ---------------------------------------------------------------------------
// Generic batched bf16 WMMA GEMM:  C[m,n] = sum_k A[m,k] * W[n,k]  (+bias)
// A: [M,K] bf16 row-major, W: [N,K] bf16 row-major (i.e. x @ W.T)
// Both operands are K-contiguous -> pure global_load_b128 fragment loads.
// 128 threads = 4 waves (2x2), 64x64 block tile, 2x2 16x16 subtiles per wave.
// Requires M%64==0, N%64==0, K%64==0 (true for every GEMM here).
// Epilogue fully specialized at compile time via template flags.
// ---------------------------------------------------------------------------
enum : unsigned {
  G_BIAS = 1u, G_RELU = 2u, G_ACCUM = 4u, G_OUTF = 8u, G_OUTB = 16u
};

template <unsigned F>
__global__ __launch_bounds__(128) void gemm_bf16_wmma(
    const bf16_t* __restrict__ A, long long strideA,
    const bf16_t* __restrict__ W, long long strideW,
    float* __restrict__ Cf, bf16_t* __restrict__ Cb, long long strideC,
    const float* __restrict__ bias,
    int M, int N, int K)
{
  const int bz = blockIdx.z;
  A += (long long)bz * strideA;
  W += (long long)bz * strideW;
  const long long coff = (long long)bz * strideC;

  const int lane = threadIdx.x & 31;
  const int wave = threadIdx.x >> 5;       // 0..3
  const int wm   = wave >> 1;              // 0..1
  const int wn   = wave & 1;               // 0..1
  const int lrow = lane & 15;
  const int hi   = lane >> 4;              // 0: lanes 0-15, 1: lanes 16-31

  const int m0 = blockIdx.y * 64 + wm * 32;
  const int n0 = blockIdx.x * 64 + wn * 32;

  // Per-lane base pointers (K-contiguous fragments; see VGPR layout tables):
  // A operand 16x32: lanes 0-15 row m, K {k0..k0+7, k0+16..k0+23};
  //                  lanes 16-31 row m, K {k0+8..15, k0+24..31}
  // B operand 32x16: lane n holds W row n; lanes 0-15 K {k0..k0+7, k0+16..k0+23},
  //                  lanes 16-31 K {k0+8..15, k0+24..31}
  const bf16_t* pa0 = A + (long long)(m0 + lrow)      * K + (hi << 3);
  const bf16_t* pa1 = A + (long long)(m0 + 16 + lrow) * K + (hi << 3);
  const bf16_t* pb0 = W + (long long)(n0 + lrow)      * K + (hi << 3);
  const bf16_t* pb1 = W + (long long)(n0 + 16 + lrow) * K + (hi << 3);

  f32x8 acc[2][2];
#pragma unroll
  for (int i = 0; i < 2; ++i)
#pragma unroll
    for (int j = 0; j < 2; ++j)
#pragma unroll
      for (int r = 0; r < 8; ++r) acc[i][j][r] = 0.0f;

#pragma unroll 2
  for (int k0 = 0; k0 < K; k0 += 32) {
    // prefetch 4 K blocks (128 elements) ahead (speculative, lowers to global_prefetch_b8)
    __builtin_prefetch(pa0 + k0 + 128, 0, 1);
    __builtin_prefetch(pb0 + k0 + 128, 0, 1);

    bf16x16 af0 = ld_frag(pa0 + k0, pa0 + k0 + 16);
    bf16x16 af1 = ld_frag(pa1 + k0, pa1 + k0 + 16);
    bf16x16 bf0 = ld_frag(pb0 + k0, pb0 + k0 + 16);
    bf16x16 bf1 = ld_frag(pb1 + k0, pb1 + k0 + 16);

    acc[0][0] = __builtin_amdgcn_wmma_f32_16x16x32_bf16(false, af0, false, bf0, (short)0, acc[0][0], false, false);
    acc[0][1] = __builtin_amdgcn_wmma_f32_16x16x32_bf16(false, af0, false, bf1, (short)0, acc[0][1], false, false);
    acc[1][0] = __builtin_amdgcn_wmma_f32_16x16x32_bf16(false, af1, false, bf0, (short)0, acc[1][0], false, false);
    acc[1][1] = __builtin_amdgcn_wmma_f32_16x16x32_bf16(false, af1, false, bf1, (short)0, acc[1][1], false, false);
  }

  // Epilogue: C/D layout — lane<16: col n0+16j+lane, rows m0+16i+r;
  //           lanes 16-31: same cols, rows +8
#pragma unroll
  for (int i = 0; i < 2; ++i)
#pragma unroll
    for (int j = 0; j < 2; ++j) {
      const int col  = n0 + 16 * j + lrow;
      const int row0 = m0 + 16 * i + (hi << 3);
      float bv = 0.0f;
      if constexpr ((F & G_BIAS) != 0) bv = bias[col];
#pragma unroll
      for (int r = 0; r < 8; ++r) {
        long long idx = coff + (long long)(row0 + r) * N + col;
        float v = acc[i][j][r] + bv;
        if constexpr ((F & G_ACCUM) != 0) v += Cf[idx];
        if constexpr ((F & G_RELU)  != 0) v = fmaxf(v, 0.0f);
        if constexpr ((F & G_OUTF)  != 0) Cf[idx] = v;
        if constexpr ((F & G_OUTB)  != 0) Cb[idx] = f2bf(v);
      }
    }
}

// ---------------------------------------------------------------------------
// qkv [M=1024, 3072] -> q,k [B,H,S,hd] and vT [B,H,hd,S]
// ---------------------------------------------------------------------------
__global__ void reorder_qkv(const bf16_t* __restrict__ qkv,
                            bf16_t* __restrict__ q, bf16_t* __restrict__ k,
                            bf16_t* __restrict__ vT) {
  int i = blockIdx.x * blockDim.x + threadIdx.x;       // 1024*3072
  if (i >= 1024 * 3072) return;
  int m = i / 3072, n = i % 3072;
  int b = m >> 9, s = m & 511;
  int part = n >> 10, f = n & 1023;
  int h = f >> 6, d = f & 63;
  bf16_t v = qkv[i];
  if (part == 0)      q [(((b * 16 + h) * 512 + s) << 6) + d] = v;
  else if (part == 1) k [(((b * 16 + h) * 512 + s) << 6) + d] = v;
  else                vT[(((b * 16 + h) * 64  + d) << 9) + s] = v;
}

// ctx_acc [B,H,S,hd] f32 -> ctx_bf [B,S, H*hd] bf16
__global__ void reorder_ctx(const float* __restrict__ ctx_acc, bf16_t* __restrict__ ctx_bf) {
  int i = blockIdx.x * blockDim.x + threadIdx.x;       // 1024*1024
  if (i >= 1024 * 1024) return;
  int d = i & 63, s = (i >> 6) & 511, h = (i >> 15) & 15, b = i >> 19;
  ctx_bf[((b * 512 + s) << 10) + (h << 6) + d] = f2bf(ctx_acc[i]);
}

// ---------------------------------------------------------------------------
// Fused relation-bias + softmax + Arel accumulation. One block per (b,h,q) row.
// scores: qk [BH,S,S] f32; P: [BH,S,64] f32; relation: [B,S,S] i32.
// Writes attn (bf16) and Arel[BH,S,64] (bf16). Scale 1/sqrt(64)=0.125 applied.
// ---------------------------------------------------------------------------
__global__ __launch_bounds__(256) void softmax_rel(
    const float* __restrict__ scores, const float* __restrict__ P,
    const int* __restrict__ relation,
    bf16_t* __restrict__ attn, bf16_t* __restrict__ Arel)
{
  const int S = 512;
  int idx = blockIdx.x;                 // (b*16+h)*512 + q
  int q = idx & 511;
  int b = idx >> 13;
  const float* srow = scores + (long long)idx * S;
  const float* Prow = P + (long long)idx * 64;
  const int*   rrow = relation + ((long long)(b * S + q)) * S;

  __shared__ float red[256];
  __shared__ float sP[64];
  __shared__ float arel[64];
  int t = threadIdx.x;
  if (t < 64) { sP[t] = Prow[t]; arel[t] = 0.0f; }
  __syncthreads();

  int r0 = rrow[t], r1 = rrow[t + 256];
  float v0 = (srow[t]       + (r0 ? sP[r0] : 0.0f)) * 0.125f;
  float v1 = (srow[t + 256] + (r1 ? sP[r1] : 0.0f)) * 0.125f;

  red[t] = fmaxf(v0, v1); __syncthreads();
  for (int o = 128; o > 0; o >>= 1) { if (t < o) red[t] = fmaxf(red[t], red[t + o]); __syncthreads(); }
  float mx = red[0]; __syncthreads();

  float e0 = __expf(v0 - mx), e1 = __expf(v1 - mx);
  red[t] = e0 + e1; __syncthreads();
  for (int o = 128; o > 0; o >>= 1) { if (t < o) red[t] += red[t + o]; __syncthreads(); }
  float inv = 1.0f / red[0];

  float a0 = e0 * inv, a1 = e1 * inv;
  attn[(long long)idx * S + t]       = f2bf(a0);
  attn[(long long)idx * S + t + 256] = f2bf(a1);
  if (r0) atomicAdd(&arel[r0], a0);
  if (r1) atomicAdd(&arel[r1], a1);
  __syncthreads();
  if (t < 64) Arel[(long long)idx * 64 + t] = f2bf(arel[t]);
}

// ---------------------------------------------------------------------------
// x = LayerNorm(a + b) * g + beta ; one block per row of 1024
// ---------------------------------------------------------------------------
template <bool WRITE_BF>
__global__ __launch_bounds__(256) void ln_residual(
    const float* __restrict__ a, const float* __restrict__ bsrc,
    const float* __restrict__ g, const float* __restrict__ beta,
    float* __restrict__ outf, bf16_t* __restrict__ outb)
{
  const int D = 1024;
  int row = blockIdx.x, t = threadIdx.x;
  const float* pa = a + (long long)row * D;
  const float* pb = bsrc + (long long)row * D;
  float v[4], s = 0.0f, ss = 0.0f;
#pragma unroll
  for (int i = 0; i < 4; ++i) {
    int c = t + i * 256;
    v[i] = pa[c] + pb[c];
    s += v[i]; ss += v[i] * v[i];
  }
  __shared__ float r1[256], r2[256];
  r1[t] = s; r2[t] = ss; __syncthreads();
  for (int o = 128; o > 0; o >>= 1) {
    if (t < o) { r1[t] += r1[t + o]; r2[t] += r2[t + o]; }
    __syncthreads();
  }
  float mean = r1[0] * (1.0f / D);
  float var  = r2[0] * (1.0f / D) - mean * mean;
  float rstd = rsqrtf(var + 1e-5f);
#pragma unroll
  for (int i = 0; i < 4; ++i) {
    int c = t + i * 256;
    float y = (v[i] - mean) * rstd * g[c] + beta[c];
    outf[(long long)row * D + c] = y;
    if constexpr (WRITE_BF) outb[(long long)row * D + c] = f2bf(y);
  }
}

// ---------------------------------------------------------------------------
// Host-side orchestration
// ---------------------------------------------------------------------------
extern "C" void kernel_launch(void* const* d_in, const int* in_sizes, int n_in,
                              void* d_out, int out_size, void* d_ws, size_t ws_size,
                              hipStream_t stream) {
  (void)in_sizes; (void)n_in; (void)out_size; (void)ws_size;

  const float* src      = (const float*)d_in[0];
  const int*   relation = (const int*)  d_in[1];
  const float* Wqkv     = (const float*)d_in[2];
  const float* bqkv     = (const float*)d_in[3];
  const float* Wo       = (const float*)d_in[4];
  const float* bo       = (const float*)d_in[5];
  const float* ln1_g    = (const float*)d_in[6];
  const float* ln1_b    = (const float*)d_in[7];
  const float* W1       = (const float*)d_in[8];
  const float* b1       = (const float*)d_in[9];
  const float* W2       = (const float*)d_in[10];
  const float* b2       = (const float*)d_in[11];
  const float* ln2_g    = (const float*)d_in[12];
  const float* ln2_b    = (const float*)d_in[13];
  const float* relk     = (const float*)d_in[14];
  const float* relv     = (const float*)d_in[15];
  float* out = (float*)d_out;

  const int M = 1024, D = 1024, DFF = 4096, S = 512, BH = 32, HD = 64;

  // bump-allocate workspace (256B aligned)
  char* base = (char*)d_ws;
  size_t off = 0;
  auto alloc = [&](size_t bytes) -> void* {
    void* p = base + off;
    off = (off + bytes + 255) & ~(size_t)255;
    return p;
  };
  bf16_t* src_bf  = (bf16_t*)alloc((size_t)M * D * 2);
  bf16_t* Wqkv_bf = (bf16_t*)alloc((size_t)3 * D * D * 2);
  bf16_t* Wo_bf   = (bf16_t*)alloc((size_t)D * D * 2);
  bf16_t* W1_bf   = (bf16_t*)alloc((size_t)DFF * D * 2);
  bf16_t* W2_bf   = (bf16_t*)alloc((size_t)D * DFF * 2);
  bf16_t* relk_bf = (bf16_t*)alloc(64 * 64 * 2);
  bf16_t* relvT   = (bf16_t*)alloc(64 * 64 * 2);
  bf16_t* qkv_bf  = (bf16_t*)alloc((size_t)M * 3 * D * 2);
  bf16_t* q_bf    = (bf16_t*)alloc((size_t)BH * S * HD * 2);
  bf16_t* k_bf    = (bf16_t*)alloc((size_t)BH * S * HD * 2);
  bf16_t* vT_bf   = (bf16_t*)alloc((size_t)BH * HD * S * 2);
  float*  Pmat    = (float*) alloc((size_t)BH * S * 64 * 4);
  float*  scores  = (float*) alloc((size_t)BH * S * S * 4);
  bf16_t* attn_bf = (bf16_t*)alloc((size_t)BH * S * S * 2);
  bf16_t* Arel_bf = (bf16_t*)alloc((size_t)BH * S * 64 * 2);
  float*  ctx_acc = (float*) alloc((size_t)BH * S * HD * 4);
  bf16_t* ctx_bf  = (bf16_t*)alloc((size_t)M * D * 2);
  float*  attn_o  = (float*) alloc((size_t)M * D * 4);
  float*  x1f     = (float*) alloc((size_t)M * D * 4);
  bf16_t* x1b     = (bf16_t*)alloc((size_t)M * D * 2);
  bf16_t* h1_bf   = (bf16_t*)alloc((size_t)M * DFF * 2);
  float*  ff      = (float*) alloc((size_t)M * D * 4);

  auto cvt = [&](const float* in, bf16_t* o, int n) {
    cvt_f32_bf16<<<(n + 255) / 256, 256, 0, stream>>>(in, o, n);
  };

  // 1) fp32 -> bf16 conversions
  cvt(src, src_bf, M * D);
  cvt(Wqkv, Wqkv_bf, 3 * D * D);
  cvt(Wo, Wo_bf, D * D);
  cvt(W1, W1_bf, DFF * D);
  cvt(W2, W2_bf, D * DFF);
  cvt(relk, relk_bf, 64 * 64);
  cvt_transpose64<<<16, 256, 0, stream>>>(relv, relvT);

  // 2) QKV projection: [1024,1024] @ Wqkv.T -> [1024,3072] bf16
  gemm_bf16_wmma<G_BIAS | G_OUTB><<<dim3(3 * D / 64, M / 64, 1), 128, 0, stream>>>(
      src_bf, 0, Wqkv_bf, 0, nullptr, qkv_bf, 0, bqkv, M, 3 * D, D);
  reorder_qkv<<<(M * 3 * D + 255) / 256, 256, 0, stream>>>(qkv_bf, q_bf, k_bf, vT_bf);

  // 3) P[bh,q,r] = q . rel_k_emb[r]  (batched, W shared)
  gemm_bf16_wmma<G_OUTF><<<dim3(1, S / 64, BH), 128, 0, stream>>>(
      q_bf, (long long)S * HD, relk_bf, 0, Pmat, nullptr, (long long)S * 64,
      nullptr, S, 64, HD);

  // 4) scores = q @ k.T (batched over B*H)
  gemm_bf16_wmma<G_OUTF><<<dim3(S / 64, S / 64, BH), 128, 0, stream>>>(
      q_bf, (long long)S * HD, k_bf, (long long)S * HD, scores, nullptr,
      (long long)S * S, nullptr, S, S, HD);

  // 5) fused rel-bias gather + scale + softmax + Arel accumulation
  softmax_rel<<<BH * S, 256, 0, stream>>>(scores, Pmat, relation, attn_bf, Arel_bf);

  // 6) ctx = attn @ V  (vT stored [hd,S] per head = B-operand layout)
  gemm_bf16_wmma<G_OUTF><<<dim3(1, S / 64, BH), 128, 0, stream>>>(
      attn_bf, (long long)S * S, vT_bf, (long long)HD * S, ctx_acc, nullptr,
      (long long)S * HD, nullptr, S, HD, S);
  // 7) ctx += Arel @ rel_v_emb  (relvT is [hd,64] = B-operand layout)
  gemm_bf16_wmma<G_OUTF | G_ACCUM><<<dim3(1, S / 64, BH), 128, 0, stream>>>(
      Arel_bf, (long long)S * 64, relvT, 0, ctx_acc, nullptr,
      (long long)S * HD, nullptr, S, HD, 64);
  reorder_ctx<<<(M * D + 255) / 256, 256, 0, stream>>>(ctx_acc, ctx_bf);

  // 8) attention output projection
  gemm_bf16_wmma<G_BIAS | G_OUTF><<<dim3(D / 64, M / 64, 1), 128, 0, stream>>>(
      ctx_bf, 0, Wo_bf, 0, attn_o, nullptr, 0, bo, M, D, D);

  // 9) x1 = LN(src + attn_out)
  ln_residual<true><<<M, 256, 0, stream>>>(src, attn_o, ln1_g, ln1_b, x1f, x1b);

  // 10) FFN: h1 = relu(x1 @ W1.T + b1)
  gemm_bf16_wmma<G_BIAS | G_RELU | G_OUTB><<<dim3(DFF / 64, M / 64, 1), 128, 0, stream>>>(
      x1b, 0, W1_bf, 0, nullptr, h1_bf, 0, b1, M, DFF, D);
  // 11) ff = h1 @ W2.T + b2
  gemm_bf16_wmma<G_BIAS | G_OUTF><<<dim3(D / 64, M / 64, 1), 128, 0, stream>>>(
      h1_bf, 0, W2_bf, 0, ff, nullptr, 0, b2, M, D, DFF);

  // 12) out = LN(x1 + ff)
  ln_residual<false><<<M, 256, 0, stream>>>(x1f, ff, ln2_g, ln2_b, out, nullptr);
}
